// MultiScaleRetention_7653631722044
// MI455X (gfx1250) — matrix operand order repaired and checked by reading, not compiled, not run on hardware
//
#include <hip/hip_runtime.h>
#include <hip/hip_bf16.h>
#include <stdint.h>

// MultiScaleRetention forward for MI455X (gfx1250, wave32, WMMA, async-LDS).
// B=2, N=4096, D_MODEL=2048, H=4, DK=256, DV=512, CHUNK=64.

typedef __attribute__((ext_vector_type(16))) __bf16 v16bf;
typedef __attribute__((ext_vector_type(8)))  __bf16 v8bf;
typedef __attribute__((ext_vector_type(8)))  float  v8f;
typedef __attribute__((ext_vector_type(4)))  int    v4i;

#define DEVINL __device__ __forceinline__

#if __has_builtin(__builtin_amdgcn_global_load_async_to_lds_b128)
#define HAS_ASYNC_LDS 1
#else
#define HAS_ASYNC_LDS 0
#endif

#define AS1 __attribute__((address_space(1)))
#define AS3 __attribute__((address_space(3)))

// 16-byte global -> LDS copy. Async (ASYNCcnt-tracked) on CDNA5.
DEVINL void copy16_g2l(void* lds, const void* g) {
#if HAS_ASYNC_LDS
  __builtin_amdgcn_global_load_async_to_lds_b128((AS1 v4i*)g, (AS3 v4i*)lds, 0, 0);
#else
  *(v8bf*)lds = *(const v8bf*)g;
#endif
}

#if HAS_ASYNC_LDS
#if __has_builtin(__builtin_amdgcn_s_wait_asynccnt)
#define WAIT_ASYNC(n) __builtin_amdgcn_s_wait_asynccnt(n)
#else
#define WAIT_ASYNC(n) asm volatile("s_wait_asynccnt %0" ::"i"(n) : "memory")
#endif
#else
#define WAIT_ASYNC(n) ((void)0)
#endif

DEVINL v8f wmma_bf16(v16bf a, v16bf b, v8f c) {
  // D = A(16x32) * B(32x16) + C(16x16), fp32 accumulate
  return __builtin_amdgcn_wmma_f32_16x16x32_bf16(false, a, false, b, (short)0, c,
                                                 false, false);
}

// A fragment: 16x32 bf16 tile, rows contiguous in LDS with row stride `ld`.
DEVINL v16bf frag_a(const __bf16* base, int ld) {
  int lane = threadIdx.x & 31;
  int m = lane & 15;
  int kb = (lane >> 4) << 3;
  const __bf16* p = base + m * ld + kb;
  v8bf lo = *(const v8bf*)(p);
  v8bf hi = *(const v8bf*)(p + 16);
  return __builtin_shufflevector(lo, hi, 0,1,2,3,4,5,6,7,8,9,10,11,12,13,14,15);
}

// B fragment: 32x16 bf16 tile stored transposed in LDS as [n][k] (row stride `ld`).
DEVINL v16bf frag_b(const __bf16* base, int ld) {
  int lane = threadIdx.x & 31;
  int n = lane & 15;
  int kb = (lane >> 4) << 3;
  const __bf16* p = base + n * ld + kb;
  v8bf lo = *(const v8bf*)(p);
  v8bf hi = *(const v8bf*)(p + 16);
  return __builtin_shufflevector(lo, hi, 0,1,2,3,4,5,6,7,8,9,10,11,12,13,14,15);
}

// ---------------- fp32 -> bf16 cast ----------------
__global__ void cast_kernel(const float* __restrict__ in, __bf16* __restrict__ out, int n) {
  int i = (blockIdx.x * 256 + threadIdx.x) * 4;
  if (i + 3 < n) {
    float4 v = *(const float4*)(in + i);
    out[i+0] = (__bf16)v.x; out[i+1] = (__bf16)v.y;
    out[i+2] = (__bf16)v.z; out[i+3] = (__bf16)v.w;
  }
}

// ---------------- GEMM: out[M,N] = A[M,K](bf16) * W[N,K]^T (bf16) ----------------
// Block: 256 thr (8 waves as 4x2), block tile 128M x 128N, wave tile 32x64.
// Triple-buffered LDS; steady-state loop is branch-free: unconditional
// s_wait_asynccnt 4 keeps exactly one prefetch stage (4 ops) in flight.
// Requires K >= 96 (nsteps >= 3); all call sites use K=2048.
// epi: 0 = bf16 raw, 1 = silu->bf16, 2 = fp32
__global__ void gemm_bf16(const __bf16* __restrict__ A, const __bf16* __restrict__ Bw,
                          __bf16* __restrict__ outb, float* __restrict__ outf,
                          int M, int N, int K, int epi) {
  __shared__ __bf16 As[3][128 * 32];   // 3 x 8 KB
  __shared__ __bf16 Bs[3][128 * 32];   // 3 x 8 KB
  const int tid = threadIdx.x;
  const int wave = tid >> 5, lane = tid & 31;
  const int wm = wave >> 1, wn = wave & 1;
  const int m0 = blockIdx.y * 128, n0 = blockIdx.x * 128;

  // tile loader: A 128x32 + W 128x32, 4 async b128 per thread per step
  auto load_tiles = [&](int buf, int k0) {
    int e0 = tid * 8;
    int r0 = e0 >> 5, c0 = e0 & 31;
    copy16_g2l(&As[buf][e0],        &A[(size_t)(m0 + r0) * K + k0 + c0]);
    copy16_g2l(&As[buf][e0 + 2048], &A[(size_t)(m0 + r0 + 64) * K + k0 + c0]);
    copy16_g2l(&Bs[buf][e0],        &Bw[(size_t)(n0 + r0) * K + k0 + c0]);
    copy16_g2l(&Bs[buf][e0 + 2048], &Bw[(size_t)(n0 + r0 + 64) * K + k0 + c0]);
  };

  v8f acc[2][4] = {};
  auto compute_step = [&](int buf) {
    v16bf a0 = frag_a(&As[buf][(wm * 32) * 32], 32);
    v16bf a1 = frag_a(&As[buf][(wm * 32 + 16) * 32], 32);
#pragma unroll
    for (int nt = 0; nt < 4; ++nt) {
      v16bf b = frag_b(&Bs[buf][(wn * 64 + nt * 16) * 32], 32);
      acc[0][nt] = wmma_bf16(a0, b, acc[0][nt]);
      acc[1][nt] = wmma_bf16(a1, b, acc[1][nt]);
    }
  };

  const int nsteps = K >> 5;           // >= 3
  load_tiles(0, 0);
  load_tiles(1, 32);

  int cur = 0, nxt2 = 2;
  for (int i = 0; i + 2 < nsteps; ++i) {
    WAIT_ASYNC(4);                     // step i complete; step i+1 may pend
    __syncthreads();
    compute_step(cur);
    load_tiles(nxt2, (i + 2) * 32);    // buffer last read at step i-1 (barrier-ordered)
    cur = (cur == 2) ? 0 : cur + 1;
    nxt2 = (nxt2 == 2) ? 0 : nxt2 + 1;
  }
  WAIT_ASYNC(4);                       // step nsteps-2 complete
  __syncthreads();
  compute_step(cur);
  cur = (cur == 2) ? 0 : cur + 1;
  WAIT_ASYNC(0);                       // final step complete
  __syncthreads();
  compute_step(cur);

  const int n = lane & 15, mb = (lane >> 4) << 3;
#pragma unroll
  for (int mt = 0; mt < 2; ++mt) {
#pragma unroll
    for (int nt = 0; nt < 4; ++nt) {
      int gm0 = m0 + wm * 32 + mt * 16;
      int gn = n0 + wn * 64 + nt * 16 + n;
#pragma unroll
      for (int r = 0; r < 8; ++r) {
        int gm = gm0 + r + mb;
        float v = acc[mt][nt][r];
        if (epi == 2) {
          outf[(size_t)gm * N + gn] = v;
        } else {
          if (epi == 1) v = v / (1.f + __expf(-v));   // silu
          outb[(size_t)gm * N + gn] = (__bf16)v;
        }
      }
    }
  }
}

// ---------------- RoPE in-place on q,k (q also scaled by dk^-0.5) ----------------
__global__ void rope_kernel(__bf16* __restrict__ q, __bf16* __restrict__ k) {
  int idx = blockIdx.x * 256 + threadIdx.x;   // B*N*H*128 threads
  int d2 = idx & 127;
  int hh = (idx >> 7) & 3;
  int m  = idx >> 9;                           // 0..8191 (b*N+n)
  int n  = m & 4095;
  float inv = exp2f(-(float)d2 * (13.287712379549449f / 128.f));  // 10000^(-2*d2/256)
  float fr = (float)n * inv;
  float s, c;
  __sincosf(fr, &s, &c);
  size_t b0 = (size_t)m * 1024 + hh * 256 + d2;
  float q1 = (float)q[b0], q2 = (float)q[b0 + 128];
  float k1 = (float)k[b0], k2 = (float)k[b0 + 128];
  q[b0]       = (__bf16)((q1 * c - q2 * s) * 0.0625f);
  q[b0 + 128] = (__bf16)((q2 * c + q1 * s) * 0.0625f);
  k[b0]       = (__bf16)(k1 * c - k2 * s);
  k[b0 + 128] = (__bf16)(k2 * c + k1 * s);
}

// ---------------- Retention chunk scan ----------------
// grid: 64 blocks = (b:2) x (h:4) x (v-slice:8 of 64 cols); block: 256 threads (8 waves).
// Dynamic LDS (272 KB): S f32[256][64] | qs[2][64][256] | ks[2][64][256] |
//   kts[256][64] | vts[64][64] | attns[64][64] | Sbt bf16[64][256]
// q/k staging is double-buffered across chunks: chunk c+1 streams in (async)
// while chunk c computes; s_wait_asynccnt 16 leaves only that prefetch pending.
__global__ void retention_kernel(const __bf16* __restrict__ qg, const __bf16* __restrict__ kg,
                                 const __bf16* __restrict__ vg, float* __restrict__ og) {
  extern __shared__ unsigned char smem[];
  float*  S     = (float*)smem;                            // 65536 B
  __bf16* qs0   = (__bf16*)(smem + 65536);                 // 2 x 32768 B
  __bf16* ks0   = qs0 + 2 * 64 * 256;                      // 2 x 32768 B
  __bf16* kts   = ks0 + 2 * 64 * 256;                      // 32768 B
  __bf16* vts   = kts + 256 * 64;                          // 8192 B
  __bf16* attns = vts + 64 * 64;                           // 8192 B
  __bf16* Sbt   = attns + 64 * 64;                         // 32768 B  [v][d]

  const int tid = threadIdx.x, lane = tid & 31, wave = tid >> 5;
  const int vs = blockIdx.x & 7;
  const int h  = (blockIdx.x >> 3) & 3;
  const int bb = blockIdx.x >> 5;

  const float gamma = 1.f - exp2f(-5.f - (float)h);
  const float l2g = log2f(gamma);
  const float gC  = exp2f(64.f * l2g);

  auto load_qk = [&](int buf, int c) {   // 16 async b128 per thread
    __bf16* qd = qs0 + buf * 64 * 256;
    __bf16* kd = ks0 + buf * 64 * 256;
#pragma unroll
    for (int it = 0; it < 8; ++it) {
      int e = (tid + it * 256) * 8;
      int i = e >> 8, d = e & 255;
      size_t gidx = ((size_t)((bb * 4096 + c * 64 + i) * 4 + h)) * 256 + d;
      copy16_g2l(&qd[i * 256 + d], &qg[gidx]);
      copy16_g2l(&kd[i * 256 + d], &kg[gidx]);
    }
  };

  for (int i = tid; i < 256 * 64; i += 256) S[i] = 0.f;
  for (int i = tid; i < 64 * 256; i += 256) Sbt[i] = (__bf16)0.f;
  load_qk(0, 0);
  __syncthreads();

  for (int c = 0; c < 64; ++c) {
    const int cur = c & 1;
    const __bf16* qs = qs0 + cur * 64 * 256;
    const __bf16* ks = ks0 + cur * 64 * 256;

    // prefetch next chunk's q,k into the other buffer (overlaps this chunk's math)
    if (c + 1 < 64) load_qk(cur ^ 1, c + 1);
    // ---- v^T slice: vts[vc][j] = v[b, c*64+j, h, vs*64+vc] (transpose) ----
#pragma unroll
    for (int it = 0; it < 16; ++it) {
      int idx = tid * 16 + it;
      int vc = idx >> 6, j = idx & 63;
      size_t gidx = ((size_t)((bb * 4096 + c * 64 + j) * 4 + h)) * 512 + vs * 64 + vc;
      vts[vc * 64 + j] = vg[gidx];
    }
    // chunk c's q/k complete; chunk c+1's 16 prefetch ops may stay in flight
    if (c + 1 < 64) { WAIT_ASYNC(16); } else { WAIT_ASYNC(0); }
    __syncthreads();
    // ---- kts[d][j] = k[j][d] * gamma^(63-j) ----
    for (int idx = tid; idx < 256 * 64; idx += 256) {
      int d = idx >> 6, j = idx & 63;
      float gk = exp2f((float)(63 - j) * l2g);
      kts[d * 64 + j] = (__bf16)((float)ks[j * 256 + d] * gk);
    }
    __syncthreads();

    v8f acc[2];
    // ---- phase A: attn tiles (masked/decayed) + o_inter = q @ S_old ----
#pragma unroll
    for (int t = 0; t < 2; ++t) {
      int tile = wave * 2 + t;
      int mi = tile >> 2, ni = tile & 3;
      v8f aa = {};
#pragma unroll
      for (int kd = 0; kd < 8; ++kd) {
        v16bf a = frag_a(qs + mi * 16 * 256 + kd * 32, 256);
        v16bf b = frag_b(ks + ni * 16 * 256 + kd * 32, 256);
        aa = wmma_bf16(a, b, aa);
      }
      int n = lane & 15, mb = (lane >> 4) << 3;
#pragma unroll
      for (int r = 0; r < 8; ++r) {
        int i = mi * 16 + r + mb, j = ni * 16 + n;
        float v = (i >= j) ? aa[r] * exp2f((float)(i - j) * l2g) : 0.f;
        attns[i * 64 + j] = (__bf16)v;
      }
      // o_inter: B tile from bf16 transposed state mirror (vector LDS loads)
      v8f oo = {};
#pragma unroll
      for (int kd = 0; kd < 8; ++kd) {
        v16bf a = frag_a(qs + mi * 16 * 256 + kd * 32, 256);
        v16bf b = frag_b(Sbt + (ni * 16) * 256 + kd * 32, 256);
        oo = wmma_bf16(a, b, oo);
      }
#pragma unroll
      for (int r = 0; r < 8; ++r) {
        int i = mi * 16 + r + mb;
        oo[r] *= exp2f((float)(i + 1) * l2g);   // g_in
      }
      acc[t] = oo;
    }
    __syncthreads();   // attn written; all reads of S/Sbt done

    // ---- phase B: o = o_inter + attn @ v; store fp32 ----
#pragma unroll
    for (int t = 0; t < 2; ++t) {
      int tile = wave * 2 + t;
      int mi = tile >> 2, ni = tile & 3;
      v8f oo = acc[t];
#pragma unroll
      for (int kd = 0; kd < 2; ++kd) {
        v16bf a = frag_a(attns + mi * 16 * 64 + kd * 32, 64);
        v16bf b = frag_b(vts + ni * 16 * 64 + kd * 32, 64);
        oo = wmma_bf16(a, b, oo);
      }
      int n = lane & 15, mb = (lane >> 4) << 3;
#pragma unroll
      for (int r = 0; r < 8; ++r) {
        int i = mi * 16 + r + mb;
        size_t gidx = ((size_t)((bb * 4096 + c * 64 + i) * 4 + h)) * 512
                      + vs * 64 + ni * 16 + n;
        og[gidx] = oo[r];
      }
    }

    // ---- phase C: S = S*gamma^C + (k*g_k)^T @ v; refresh bf16 mirror ----
#pragma unroll
    for (int t = 0; t < 8; ++t) {
      int tile = wave * 8 + t;
      int di = tile >> 2, vi = tile & 3;
      int n = lane & 15, mb = (lane >> 4) << 3;
      int col = vi * 16 + n;
      v8f cf;
#pragma unroll
      for (int r = 0; r < 8; ++r)
        cf[r] = S[(di * 16 + r + mb) * 64 + col] * gC;
#pragma unroll
      for (int kd = 0; kd < 2; ++kd) {
        v16bf a = frag_a(kts + di * 16 * 64 + kd * 32, 64);
        v16bf b = frag_b(vts + vi * 16 * 64 + kd * 32, 64);
        cf = wmma_bf16(a, b, cf);
      }
#pragma unroll
      for (int r = 0; r < 8; ++r) {
        int d = di * 16 + r + mb;
        S[d * 64 + col] = cf[r];
        Sbt[col * 256 + d] = (__bf16)cf[r];
      }
    }
    __syncthreads();
  }
}

// ---------------- group RMS-norm * gnorm_w * silu-gate -> bf16 ----------------
__global__ void normgate_kernel(const float* __restrict__ o, const __bf16* __restrict__ gb,
                                const float* __restrict__ gnw, __bf16* __restrict__ ogb) {
  int wave = threadIdx.x >> 5, lane = threadIdx.x & 31;
  int wid = blockIdx.x * 8 + wave;       // (m,h), m=b*N+n
  int hh = wid & 3, m = wid >> 2;
  size_t base = (size_t)(m * 4 + hh) * 512;
  float ss = 0.f;
#pragma unroll
  for (int t = 0; t < 16; ++t) { float v = o[base + lane + t * 32]; ss += v * v; }
#pragma unroll
  for (int off = 16; off > 0; off >>= 1) ss += __shfl_xor(ss, off, 32);
  float rms = rsqrtf(ss * (1.f / 512.f) + 1e-5f);
  size_t ob = (size_t)m * 2048 + hh * 512;
#pragma unroll
  for (int t = 0; t < 16; ++t) {
    int dv = lane + t * 32;
    float on = o[base + dv] * rms * gnw[dv];
    float gt = (float)gb[ob + dv];       // gate already silu'd
    ogb[ob + dv] = (__bf16)(on * gt);
  }
}

extern "C" void kernel_launch(void* const* d_in, const int* in_sizes, int n_in,
                              void* d_out, int out_size, void* d_ws, size_t ws_size,
                              hipStream_t stream) {
  (void)in_sizes; (void)n_in; (void)out_size; (void)ws_size;
  const float* x   = (const float*)d_in[0];
  const float* Wq  = (const float*)d_in[1];
  const float* Wk  = (const float*)d_in[2];
  const float* Wv  = (const float*)d_in[3];
  const float* Wg  = (const float*)d_in[4];
  const float* Wo  = (const float*)d_in[5];
  const float* gnw = (const float*)d_in[6];
  float* out = (float*)d_out;

  const int M = 2 * 4096;        // 8192 rows

  uint8_t* ws = (uint8_t*)d_ws;
  size_t off = 0;
  auto alloc = [&](size_t bytes) {
    uint8_t* p = ws + off;
    off += (bytes + 255) & ~(size_t)255;
    return p;
  };
  __bf16* xb   = (__bf16*)alloc((size_t)M * 2048 * 2);
  __bf16* wqb  = (__bf16*)alloc((size_t)1024 * 2048 * 2);
  __bf16* wkb  = (__bf16*)alloc((size_t)1024 * 2048 * 2);
  __bf16* wvb  = (__bf16*)alloc((size_t)2048 * 2048 * 2);
  __bf16* wgb  = (__bf16*)alloc((size_t)2048 * 2048 * 2);
  __bf16* wob  = (__bf16*)alloc((size_t)2048 * 2048 * 2);
  __bf16* qb   = (__bf16*)alloc((size_t)M * 1024 * 2);
  __bf16* kb   = (__bf16*)alloc((size_t)M * 1024 * 2);
  __bf16* vb   = (__bf16*)alloc((size_t)M * 2048 * 2);
  __bf16* gbuf = (__bf16*)alloc((size_t)M * 2048 * 2);
  float*  obuf = (float*)alloc((size_t)M * 2048 * 4);
  __bf16* ogb  = (__bf16*)alloc((size_t)M * 2048 * 2);

  // casts
  cast_kernel<<<(M * 2048) / 1024, 256, 0, stream>>>(x,  xb,  M * 2048);
  cast_kernel<<<(1024 * 2048) / 1024, 256, 0, stream>>>(Wq, wqb, 1024 * 2048);
  cast_kernel<<<(1024 * 2048) / 1024, 256, 0, stream>>>(Wk, wkb, 1024 * 2048);
  cast_kernel<<<(2048 * 2048) / 1024, 256, 0, stream>>>(Wv, wvb, 2048 * 2048);
  cast_kernel<<<(2048 * 2048) / 1024, 256, 0, stream>>>(Wg, wgb, 2048 * 2048);
  cast_kernel<<<(2048 * 2048) / 1024, 256, 0, stream>>>(Wo, wob, 2048 * 2048);

  // projections (block tile 128x128)
  gemm_bf16<<<dim3(1024 / 128, M / 128), 256, 0, stream>>>(xb, wqb, qb, nullptr, M, 1024, 2048, 0);
  gemm_bf16<<<dim3(1024 / 128, M / 128), 256, 0, stream>>>(xb, wkb, kb, nullptr, M, 1024, 2048, 0);
  gemm_bf16<<<dim3(2048 / 128, M / 128), 256, 0, stream>>>(xb, wvb, vb, nullptr, M, 2048, 2048, 0);
  gemm_bf16<<<dim3(2048 / 128, M / 128), 256, 0, stream>>>(xb, wgb, gbuf, nullptr, M, 2048, 2048, 1);

  // RoPE (in-place) on q,k; q scaled by dk^-0.5
  rope_kernel<<<(M * 4 * 128) / 256, 256, 0, stream>>>(qb, kb);

  // chunked retention scan: 272 KB dynamic LDS per block
  retention_kernel<<<64, 256, 278528, stream>>>(qb, kb, vb, obuf);

  // RMS-norm + gate
  normgate_kernel<<<(M * 4) / 8, 256, 0, stream>>>(obuf, gbuf, gnw, ogb);

  // output projection (fp32 out)
  gemm_bf16<<<dim3(2048 / 128, M / 128), 256, 0, stream>>>(ogb, wob, nullptr, out, M, 2048, 2048, 2);
}
